// ConfidenceHead_65292092833887
// MI455X (gfx1250) — matrix operand order repaired and checked by reading, not compiled, not run on hardware
//
#include <hip/hip_runtime.h>
#include <math.h>

// ============================================================================
// ConfidenceHead for MI455X (gfx1250, wave32, WMMA bf16 16x16x32)
//
// Input order assumption (setup_inputs dict order, params flattened in
// jax tree order = dict keys sorted alphabetically, lists in order,
// {'b','w'} -> b before w):
//   0 atom_scalar (N*512 f32)   1 atom_norms (N*192 f32)
//   2 atom_pose_ptr (P+1 i32)   3 atom_frag_id (N i32)
//   4 pose_stats (P*6 f32)
//   5 atom_head0.b  6 atom_head0.w  7 atom_head1.b  8 atom_head1.w
//   9 attn_in_b    10 attn_in_w    11 attn_out_b   12 attn_out_w
//  13 frag_head0.b 14 frag_head0.w 15 frag_head1.b 16 frag_head1.w
//  17 pool_queries 18 pose_head0.b 19 pose_head0.w 20 pose_head1.b 21 pose_head1.w
//  22 trunk0.b 23 trunk0.w 24 trunk1.b 25 trunk1.w 26 trunk2.b 27 trunk2.w
//
// d_out layout (f32, concat in return order):
//   atom_disp_log1p(N) atom_plddt(N) atom_disp(N)
//   frag_rmsd_log1p(NF) frag_bad(NF) frag_rmsd(NF)
//   pose_rmsd_log1p(P) pose_prob(P) pose_rmsd(P) pose_rmsd_from_atoms(P)
// ============================================================================

#define DEVINL __device__ __forceinline__

typedef __bf16 bf16;
typedef __attribute__((ext_vector_type(8)))  __bf16 v8bf;
typedef __attribute__((ext_vector_type(16))) __bf16 v16bf;
typedef __attribute__((ext_vector_type(8)))  float  v8f;
typedef __attribute__((ext_vector_type(4)))  int    v4i;
typedef __attribute__((address_space(1))) v4i as1_v4i;  // global 16B chunk
typedef __attribute__((address_space(3))) v4i as3_v4i;  // LDS 16B chunk

#if __has_builtin(__builtin_amdgcn_global_load_async_to_lds_b128)
#define HAVE_ASYNC_COPY 1
#else
#define HAVE_ASYNC_COPY 0
#endif

static constexpr int N_ATOM = 131072;
static constexpr int P_POSE = 1024;
static constexpr int NFRAG  = 16384;
static constexpr int HID    = 256;
static constexpr int IN_DIM = 704;   // 512 scalar + 192 norms
static constexpr int PF_DIM = 1542;  // 2*HID + 4*HID + 6
static constexpr int PF_LD  = 1568;  // padded to multiple of 32

// ---- workspace layout (bytes) ----
static constexpr size_t OFF_H2   = 0;                                      // N*HID bf16
static constexpr size_t OFF_FACC = OFF_H2   + (size_t)N_ATOM * HID * 2;    // NF*HID f32
static constexpr size_t OFF_FCNT = OFF_FACC + (size_t)NFRAG * HID * 4;     // NF f32
static constexpr size_t OFF_W0   = OFF_FCNT + (size_t)NFRAG * 4;           // 256x704 bf16
static constexpr size_t OFF_W1   = OFF_W0   + (size_t)HID * IN_DIM * 2;    // 256x256 bf16
static constexpr size_t OFF_W2   = OFF_W1   + (size_t)HID * HID * 2;
static constexpr size_t OFF_WA0  = OFF_W2   + (size_t)HID * HID * 2;
static constexpr size_t OFF_WF0  = OFF_WA0  + (size_t)HID * HID * 2;
static constexpr size_t OFF_WK   = OFF_WF0  + (size_t)HID * HID * 2;
static constexpr size_t OFF_WV   = OFF_WK   + (size_t)HID * HID * 2;
static constexpr size_t OFF_WP0  = OFF_WV   + (size_t)HID * HID * 2;       // 256x1568 bf16
static constexpr size_t OFF_QM   = OFF_WP0  + (size_t)HID * PF_LD * 2;     // 16x256 bf16
static constexpr size_t OFF_PF   = OFF_QM   + (size_t)16 * HID * 2;        // P*1568 bf16

// ---------------------------------------------------------------------------
DEVINL float silu_f(float x) { return x / (1.0f + __expf(-x)); }
DEVINL float cexpm1(float x) {
    x = fminf(fmaxf(x, -2.0f), 5.0f);
    return fmaxf(expm1f(x), 0.0f);
}

DEVINL v16bf combine16(v8bf lo, v8bf hi) {
    v16bf a;
#pragma unroll
    for (int i = 0; i < 8; ++i) { a[i] = lo[i]; a[i + 8] = hi[i]; }
    return a;
}

// A-fragment (16x32 bf16, ISA 7.12.2): lane<16 -> row=lane, K {k0..k0+7, k0+16..k0+23}
//                                      lane>=16 -> row=lane-16, K {+8, +24}
DEVINL v16bf load_frag_A(const bf16* base, int ld, int k0, int lane) {
    int row = lane & 15, half = lane >> 4;
    const bf16* p = base + (size_t)row * ld + k0 + half * 8;
    v8bf lo = *(const v8bf*)(p);
    v8bf hi = *(const v8bf*)(p + 16);
    return combine16(lo, hi);
}

// B-fragment (32x16 bf16): lane<16 -> col=n0+lane, K {k0..k0+7, k0+16..k0+23} ; lane>=16 -> K {+8, +24}
// W stored row-major [Dout][K] (col of B == row of W since we compute X @ W^T)
DEVINL v16bf load_frag_B(const bf16* W, int ldk, int n0, int k0, int lane) {
    int col = n0 + (lane & 15), half = lane >> 4;
    const bf16* p = W + (size_t)col * ldk + k0 + half * 8;
    v8bf lo = *(const v8bf*)(p);
    v8bf hi = *(const v8bf*)(p + 16);
    return combine16(lo, hi);
}

DEVINL void zero8(v8f& a) {
#pragma unroll
    for (int r = 0; r < 8; ++r) a[r] = 0.0f;
}

// ---------------------------------------------------------------------------
// 32(M) x 128(N) wave tile: 2 m-tiles x 8 n-tiles, each B fragment reused for
// two WMMAs (halves B-operand bandwidth vs 16x256 tiling).
// A: bf16 row-major (usually LDS), rows = 32. W: bf16 row-major [256][K].
// ---------------------------------------------------------------------------
DEVINL void gemm32x128(v8f acc[2][8], const bf16* A, int lda,
                       const bf16* W, int K, int n_base, int lane) {
    for (int k0 = 0; k0 < K; k0 += 32) {
        if (k0 + 32 < K)  // prefetch next weight K-slice (global_prefetch_b8)
            __builtin_prefetch(W + (size_t)(n_base + (lane & 15)) * K + k0 + 32, 0, 1);
        v16bf a0 = load_frag_A(A, lda, k0, lane);
        v16bf a1 = load_frag_A(A + (size_t)16 * lda, lda, k0, lane);
#pragma unroll
        for (int nt = 0; nt < 8; ++nt) {
            v16bf b = load_frag_B(W, K, n_base + nt * 16, k0, lane);
            acc[0][nt] = __builtin_amdgcn_wmma_f32_16x16x32_bf16(
                false, a0, false, b, (short)0, acc[0][nt], false, false);
            acc[1][nt] = __builtin_amdgcn_wmma_f32_16x16x32_bf16(
                false, a1, false, b, (short)0, acc[1][nt], false, false);
        }
    }
}

// C layout (ISA): VGPR r, lane<16: M=r,N=lane ; lane>=16: M=8+r,N=lane-16
DEVINL void epi32x128(v8f acc[2][8], const float* bias, bool do_silu,
                      bf16* out, int ldo, int row_base, int n_base, int lane) {
    int half = lane >> 4, cl = lane & 15;
#pragma unroll
    for (int mt = 0; mt < 2; ++mt)
#pragma unroll
        for (int nt = 0; nt < 8; ++nt) {
            float b = bias[n_base + nt * 16 + cl];
#pragma unroll
            for (int r = 0; r < 8; ++r) {
                float v = acc[mt][nt][r] + b;
                if (do_silu) v = silu_f(v);
                out[(size_t)(row_base + mt * 16 + r + 8 * half) * ldo
                    + n_base + nt * 16 + cl] = (bf16)v;
            }
        }
}

// 16xHID (single m-tile) variant used by the small row-count kernels.
DEVINL void gemm16x256(v8f acc[16], const bf16* A, int lda, const bf16* W, int K, int lane) {
    for (int k0 = 0; k0 < K; k0 += 32) {
        v16bf a = load_frag_A(A, lda, k0, lane);
#pragma unroll
        for (int nt = 0; nt < 16; ++nt) {
            v16bf b = load_frag_B(W, K, nt * 16, k0, lane);
            acc[nt] = __builtin_amdgcn_wmma_f32_16x16x32_bf16(
                false, a, false, b, (short)0, acc[nt], false, false);
        }
    }
}

DEVINL void epi16x256(v8f acc[16], const float* bias, bool do_silu,
                      bf16* out, int ldo, int lane) {
    int half = lane >> 4, cl = lane & 15;
#pragma unroll
    for (int nt = 0; nt < 16; ++nt) {
        float b = bias[nt * 16 + cl];
#pragma unroll
        for (int r = 0; r < 8; ++r) {
            float v = acc[nt][r] + b;
            if (do_silu) v = silu_f(v);
            out[(size_t)(r + 8 * half) * ldo + nt * 16 + cl] = (bf16)v;
        }
    }
}

// ============================================================================
// K0: weight conversion / Qmask build / fragment-accumulator zeroing
// ============================================================================
__global__ __launch_bounds__(256) void prep_kernel(
    const float* t0w, const float* t1w, const float* t2w,
    const float* a0w, const float* f0w,
    const float* attn_in_w, const float* attn_in_b, const float* poolq,
    const float* ph0w,
    bf16* W0, bf16* W1, bf16* W2, bf16* WA0, bf16* WF0,
    bf16* WK, bf16* WV, bf16* WP0, bf16* QM,
    float* facc, float* fcnt)
{
    size_t g = (size_t)blockIdx.x * blockDim.x + threadIdx.x;
    size_t stride = (size_t)gridDim.x * blockDim.x;

    for (size_t i = g; i < (size_t)HID * IN_DIM; i += stride) W0[i] = (bf16)t0w[i];
    for (size_t i = g; i < (size_t)HID * HID; i += stride) {
        W1[i]  = (bf16)t1w[i];
        W2[i]  = (bf16)t2w[i];
        WA0[i] = (bf16)a0w[i];
        WF0[i] = (bf16)f0w[i];
        WK[i]  = (bf16)attn_in_w[(size_t)HID * HID + i];       // rows 256..511
        WV[i]  = (bf16)attn_in_w[(size_t)2 * HID * HID + i];   // rows 512..767
    }
    for (size_t i = g; i < (size_t)HID * PF_LD; i += stride) {
        int d = (int)(i / PF_LD), k = (int)(i % PF_LD);
        WP0[i] = (k < PF_DIM) ? (bf16)ph0w[(size_t)d * PF_DIM + k] : (bf16)0.0f;
    }
    // Qmask[q*4+h][k] = (k in head-h slice) ? qp[q][k] : 0 ; qp = poolq @ Wq^T + bq
    for (size_t i = g; i < 16 * HID; i += stride) {
        int row = (int)(i >> 8), k = (int)(i & 255);
        int h = row & 3, q = row >> 2;
        float v = 0.0f;
        if ((k >> 6) == h) {
            float s = attn_in_b[k];
            for (int j = 0; j < HID; ++j)
                s += poolq[q * HID + j] * attn_in_w[(size_t)k * HID + j];  // Wq rows 0..255
            v = s;
        }
        QM[i] = (bf16)v;
    }
    for (size_t i = g; i < (size_t)NFRAG * HID; i += stride) facc[i] = 0.0f;
    for (size_t i = g; i < (size_t)NFRAG; i += stride) fcnt[i] = 0.0f;
}

// ============================================================================
// K1: fused trunk (3 layers) + atom head + fragment accumulation.
// Block = 128 atoms, 8 waves as 4 M-groups(32 rows) x 2 N-groups(128 cols).
// LDS: X tile 128x704 bf16 (176 KB), reused as the 128x256 activation buffer
// after layer 0.
// ============================================================================
__global__ __launch_bounds__(256) void trunk_kernel(
    const float* xs, const float* xn, const int* frag_id,
    const float* b0, const float* b1, const float* b2,
    const float* ba0, const float* ba1, const float* wa1,
    const bf16* W0, const bf16* W1, const bf16* W2, const bf16* WA0,
    bf16* h_out, float* frag_acc, float* frag_cnt,
    float* o_dlog, float* o_plddt, float* o_disp)
{
    extern __shared__ char smem_raw[];
    bf16* xbuf = (bf16*)smem_raw;   // [128][704]; aliases act [128][256] later
    bf16* act  = (bf16*)smem_raw;

    int tid = threadIdx.x, w = tid >> 5, lane = tid & 31;
    int mg = w >> 1, ng = w & 1;
    int row_base = mg * 32, n_base = ng * 128;
    int m_blk = blockIdx.x * 128;
    int half = lane >> 4, cl = lane & 15;

    // ---- stage X tile (fp32 -> bf16), pairwise so stores are u32 ----
    {
        unsigned* xb32 = (unsigned*)xbuf;
        for (int i = tid; i < 128 * (IN_DIM / 2); i += 256) {
            int row = i / (IN_DIM / 2), cp = i % (IN_DIM / 2);
            int col = cp * 2;
            int atom = m_blk + row;
            float a, b;
            if (col < 512) {
                const float* p = xs + (size_t)atom * 512 + col;
                a = p[0]; b = p[1];
            } else {
                const float* p = xn + (size_t)atom * 192 + (col - 512);
                a = p[0]; b = p[1];
            }
            union { bf16 h[2]; unsigned u; } cv;
            cv.h[0] = (bf16)a; cv.h[1] = (bf16)b;
            xb32[i] = cv.u;
        }
    }
    __syncthreads();

    v8f acc[2][8];

    // ---- trunk layer 0 (K = 704) ----
#pragma unroll
    for (int mt = 0; mt < 2; ++mt)
#pragma unroll
        for (int nt = 0; nt < 8; ++nt) zero8(acc[mt][nt]);
    gemm32x128(acc, xbuf + (size_t)row_base * IN_DIM, IN_DIM, W0, IN_DIM, n_base, lane);
    __syncthreads();  // all X reads done before act overwrites xbuf
    epi32x128(acc, b0, true, act, HID, row_base, n_base, lane);
    __syncthreads();

    // ---- trunk layer 1 ----
#pragma unroll
    for (int mt = 0; mt < 2; ++mt)
#pragma unroll
        for (int nt = 0; nt < 8; ++nt) zero8(acc[mt][nt]);
    gemm32x128(acc, act + (size_t)row_base * HID, HID, W1, HID, n_base, lane);
    __syncthreads();
    epi32x128(acc, b1, true, act, HID, row_base, n_base, lane);
    __syncthreads();

    // ---- trunk layer 2 -> h : store act + spill bf16 + fragment atomics ----
#pragma unroll
    for (int mt = 0; mt < 2; ++mt)
#pragma unroll
        for (int nt = 0; nt < 8; ++nt) zero8(acc[mt][nt]);
    gemm32x128(acc, act + (size_t)row_base * HID, HID, W2, HID, n_base, lane);
    __syncthreads();
    {
        int fids[2][8];
#pragma unroll
        for (int mt = 0; mt < 2; ++mt)
#pragma unroll
            for (int r = 0; r < 8; ++r)
                fids[mt][r] = frag_id[m_blk + row_base + mt * 16 + r + 8 * half];
#pragma unroll
        for (int mt = 0; mt < 2; ++mt)
#pragma unroll
            for (int nt = 0; nt < 8; ++nt) {
                int col = n_base + nt * 16 + cl;
                float bb = b2[col];
#pragma unroll
                for (int r = 0; r < 8; ++r) {
                    float v = silu_f(acc[mt][nt][r] + bb);
                    int row = row_base + mt * 16 + r + 8 * half;
                    act[(size_t)row * HID + col] = (bf16)v;
                    h_out[(size_t)(m_blk + row) * HID + col] = (bf16)v;
                    atomicAdd(&frag_acc[(size_t)fids[mt][r] * HID + col], v);
                }
            }
        if (ng == 0)  // one count per atom (waves with ng==0 own 32 rows each)
            atomicAdd(&frag_cnt[frag_id[m_blk + row_base + lane]], 1.0f);
    }
    __syncthreads();

    // ---- atom head layer 0 ----
#pragma unroll
    for (int mt = 0; mt < 2; ++mt)
#pragma unroll
        for (int nt = 0; nt < 8; ++nt) zero8(acc[mt][nt]);
    gemm32x128(acc, act + (size_t)row_base * HID, HID, WA0, HID, n_base, lane);
    __syncthreads();
    epi32x128(acc, ba0, true, act, HID, row_base, n_base, lane);
    __syncthreads();

    // ---- atom head layer 1: 2 outputs, VALU dot (wave w -> rows 16w..) ----
    {
        int row = w * 16 + (lane & 15), c = lane >> 4;  // c in {0,1}
        float s = ba1[c];
        for (int k = 0; k < HID; ++k)
            s += (float)act[(size_t)row * HID + k] * wa1[(size_t)c * HID + k];
        int m = m_blk + row;
        if (c == 0) { o_dlog[m] = s; o_disp[m] = cexpm1(s); }
        else        { o_plddt[m] = s; }
    }
}

// ============================================================================
// K2: fragment head. One wave = 16 fragments (small kernel, 16x256 tiling).
// ============================================================================
__global__ __launch_bounds__(256) void frag_kernel(
    const float* frag_acc, const float* frag_cnt,
    const bf16* WF0, const float* fb0, const float* fb1, const float* fw1,
    float* o_flog, float* o_fbad, float* o_frmsd)
{
    extern __shared__ char smem_raw[];
    int w = threadIdx.x >> 5, lane = threadIdx.x & 31;
    bf16* act = (bf16*)smem_raw + (size_t)w * 16 * HID;
    int f0 = blockIdx.x * 128 + w * 16;

    for (int i = lane; i < 16 * HID; i += 32) {
        int row = i >> 8, col = i & 255;
        int f = f0 + row;
        float c = fmaxf(frag_cnt[f], 1.0f);
        act[i] = (bf16)(frag_acc[(size_t)f * HID + col] / c);
    }

    v8f acc[16];
#pragma unroll
    for (int i = 0; i < 16; ++i) zero8(acc[i]);
    gemm16x256(acc, act, HID, WF0, HID, lane);
    epi16x256(acc, fb0, true, act, HID, lane);

    {
        int row = lane & 15, c = lane >> 4;
        float s = fb1[c];
        for (int k = 0; k < HID; ++k)
            s += (float)act[(size_t)row * HID + k] * fw1[(size_t)c * HID + k];
        int f = f0 + row;
        if (c == 0) { o_flog[f] = s; o_frmsd[f] = cexpm1(s); }
        else        { o_fbad[f] = s; }
    }
}

// ============================================================================
// K3: per-pose pooling + attention. One block = one pose (128 atoms).
// h tile gathered global->LDS via async-to-LDS (ASYNCcnt) when available.
// ============================================================================
__global__ __launch_bounds__(256) void pose_kernel(
    const bf16* h2, const float* disp, const int* pptr, const float* stats,
    const bf16* WK, const bf16* WV, const bf16* QM,
    const float* attn_in_b, const float* wout, const float* bout,
    bf16* pf, float* o_prfa)
{
    extern __shared__ char smem_raw[];
    bf16*  hbuf  = (bf16*)smem_raw;            // [128][256]
    bf16*  kbuf  = hbuf + 128 * 256;           // [128][256]  K row-major (s,k)
    bf16*  vtbuf = kbuf + 128 * 256;           // [256][128]  V transposed (n,s)
    float* sc    = (float*)(vtbuf + 256 * 128);// [16][128]
    bf16*  attn  = (bf16*)(sc + 16 * 128);     // [16][128]
    float* aobuf = (float*)(attn + 16 * 128);  // [4][256]
    float* red   = aobuf + 4 * 256;            // [128]

    int tid = threadIdx.x, lane = tid & 31, w = tid >> 5;
    int mg = w >> 1, ng = w & 1;
    int p = blockIdx.x;
    int start = pptr[p];
    int size  = pptr[p + 1] - start;
    if (size < 0) size = 0;
    if (size > 128) size = 128;
    float fsz = (float)(size > 0 ? size : 1);

    // ---- gather h tile (zero-padded rows beyond size) ----
#if HAVE_ASYNC_COPY
    {
        const char* h2b = (const char*)h2;
        char* hb = (char*)hbuf;
#pragma unroll
        for (int j = 0; j < 16; ++j) {
            int i = tid + 256 * j;          // 16-byte chunk id, 4096 total
            int s = i >> 5, ch = i & 31;
            int idx = start + s;
            idx = idx < 0 ? 0 : (idx > N_ATOM - 1 ? N_ATOM - 1 : idx);
            const char* gp = h2b + (size_t)idx * 512 + ch * 16;
            char* lp = hb + (size_t)s * 512 + ch * 16;
            __builtin_amdgcn_global_load_async_to_lds_b128(
                (as1_v4i*)gp, (as3_v4i*)lp, 0, 0);
        }
#if __has_builtin(__builtin_amdgcn_s_wait_asynccnt)
        __builtin_amdgcn_s_wait_asynccnt(0);
#else
        asm volatile("s_wait_asynccnt 0" ::: "memory");
#endif
        if (size < 128) {  // uniform branch: zero invalid tail rows
            unsigned* hbu = (unsigned*)hbuf;
            for (int i = tid; i < 128 * 128; i += 256)
                if ((i >> 7) >= size) hbu[i] = 0u;
        }
    }
#else
    {
        const unsigned* h2u = (const unsigned*)h2;
        unsigned* hbu = (unsigned*)hbuf;
        for (int i = tid; i < 128 * 128; i += 256) {
            int s = i >> 7;
            int idx = start + s;
            idx = idx < 0 ? 0 : (idx > N_ATOM - 1 ? N_ATOM - 1 : idx);
            hbu[i] = (s < size) ? h2u[(size_t)idx * 128 + (i & 127)] : 0u;
        }
    }
#endif
    __syncthreads();

    // ---- mean / max pool (thread = column) ----
    {
        int c = tid;
        float sum = 0.0f, mx = -INFINITY;
        for (int s = 0; s < 128; ++s)
            if (s < size) {
                float f = (float)hbuf[(size_t)s * 256 + c];
                sum += f; mx = fmaxf(mx, f);
            }
        pf[(size_t)p * PF_LD + c]       = (bf16)(sum / fsz);
        pf[(size_t)p * PF_LD + 256 + c] = (bf16)mx;
    }
    if (tid < 32) {  // pose_stats + zero pad of pose_feat tail (1536..1567)
        float v = (tid < 6) ? stats[(size_t)p * 6 + tid] : 0.0f;
        pf[(size_t)p * PF_LD + 1536 + tid] = (bf16)v;
    }
    if (tid < 128) {  // disp^2 for rmsd_from_atoms
        int s = tid, idx = start + s;
        idx = idx < 0 ? 0 : (idx > N_ATOM - 1 ? N_ATOM - 1 : idx);
        float d = (s < size) ? disp[idx] : 0.0f;
        red[tid] = d * d;
    }

    // ---- K = h @ Wk^T + bk ; V = h @ Wv^T + bv (V stored transposed) ----
    {
        int row_base = mg * 32, n_base = ng * 128;
        int half = lane >> 4, cl = lane & 15;
        v8f acc[2][8];
#pragma unroll
        for (int mt = 0; mt < 2; ++mt)
#pragma unroll
            for (int nt = 0; nt < 8; ++nt) zero8(acc[mt][nt]);
        gemm32x128(acc, hbuf + (size_t)row_base * 256, 256, WK, HID, n_base, lane);
#pragma unroll
        for (int mt = 0; mt < 2; ++mt)
#pragma unroll
            for (int nt = 0; nt < 8; ++nt) {
                float b = attn_in_b[256 + n_base + nt * 16 + cl];
#pragma unroll
                for (int r = 0; r < 8; ++r)
                    kbuf[(size_t)(row_base + mt * 16 + r + 8 * half) * 256
                         + n_base + nt * 16 + cl] = (bf16)(acc[mt][nt][r] + b);
            }
#pragma unroll
        for (int mt = 0; mt < 2; ++mt)
#pragma unroll
            for (int nt = 0; nt < 8; ++nt) zero8(acc[mt][nt]);
        gemm32x128(acc, hbuf + (size_t)row_base * 256, 256, WV, HID, n_base, lane);
#pragma unroll
        for (int mt = 0; mt < 2; ++mt)
#pragma unroll
            for (int nt = 0; nt < 8; ++nt) {
                float b = attn_in_b[512 + n_base + nt * 16 + cl];
#pragma unroll
                for (int r = 0; r < 8; ++r) {
                    int s = row_base + mt * 16 + r + 8 * half;
                    int n = n_base + nt * 16 + cl;
                    vtbuf[(size_t)n * 128 + s] = (bf16)(acc[mt][nt][r] + b);
                }
            }
    }
    __syncthreads();

    // ---- scores = Qmask(16x256) @ K^T, wave w handles s-tile w ----
    {
        v8f acc;
        zero8(acc);
        for (int k0 = 0; k0 < 256; k0 += 32) {
            v16bf a = load_frag_A(QM, 256, k0, lane);
            v16bf b = load_frag_B(kbuf, 256, w * 16, k0, lane);  // col == s
            acc = __builtin_amdgcn_wmma_f32_16x16x32_bf16(
                false, a, false, b, (short)0, acc, false, false);
        }
        int half = lane >> 4, cl = lane & 15, s = w * 16 + cl;
#pragma unroll
        for (int r = 0; r < 8; ++r) {
            int qh = r + 8 * half;
            float v = acc[r] * 0.125f;  // 1/sqrt(DH=64)
            if (s >= size) v = -1.0e9f;
            sc[(size_t)qh * 128 + s] = v;
        }
    }
    if (tid == 0) {
        float sum = 0.0f;
        for (int i = 0; i < 128; ++i) sum += red[i];
        o_prfa[p] = sqrtf(fmaxf(sum / fsz, 0.0f));
    }
    __syncthreads();

    // ---- softmax over s (16 rows, thread per row) ----
    if (tid < 16) {
        float m = -INFINITY;
        for (int s = 0; s < 128; ++s) m = fmaxf(m, sc[(size_t)tid * 128 + s]);
        float sum = 0.0f;
        for (int s = 0; s < 128; ++s) {
            float e = __expf(sc[(size_t)tid * 128 + s] - m);
            sum += e;
            sc[(size_t)tid * 128 + s] = e;
        }
        float inv = 1.0f / sum;
        for (int s = 0; s < 128; ++s)
            attn[(size_t)tid * 128 + s] = (bf16)(sc[(size_t)tid * 128 + s] * inv);
    }
    __syncthreads();

    // ---- AV = attn(16x128) @ V(128x256); keep only matching head slice ----
    {
#pragma unroll
        for (int g = 0; g < 2; ++g) {
            int n0 = (w + 8 * g) * 16;
            v8f acc;
            zero8(acc);
            for (int k0 = 0; k0 < 128; k0 += 32) {
                v16bf a = load_frag_A(attn, 128, k0, lane);
                v16bf b = load_frag_B(vtbuf, 128, n0, k0, lane);
                acc = __builtin_amdgcn_wmma_f32_16x16x32_bf16(
                    false, a, false, b, (short)0, acc, false, false);
            }
            int half = lane >> 4, cl = lane & 15;
            int col = n0 + cl, ho = col >> 6;
#pragma unroll
            for (int r = 0; r < 8; ++r) {
                int qh = r + 8 * half;                // row = q*4 + h
                if ((qh & 3) == ho)
                    aobuf[(size_t)(qh >> 2) * 256 + col] = acc[r];
            }
        }
    }
    __syncthreads();

    // ---- ao @ attn_out_w^T + b -> attn_pool (thread = out column, 4 queries) ----
    {
        int c = tid;
        float a0 = bout[c], a1 = a0, a2 = a0, a3 = a0;
        for (int k = 0; k < 256; ++k) {
            float wv = wout[(size_t)c * 256 + k];
            a0 += aobuf[k] * wv;
            a1 += aobuf[256 + k] * wv;
            a2 += aobuf[512 + k] * wv;
            a3 += aobuf[768 + k] * wv;
        }
        size_t base = (size_t)p * PF_LD + 512;
        pf[base + c]       = (bf16)a0;
        pf[base + 256 + c] = (bf16)a1;
        pf[base + 512 + c] = (bf16)a2;
        pf[base + 768 + c] = (bf16)a3;
    }
}

// ============================================================================
// K4: pose head over pose_feat (P x 1568 bf16). One wave = 16 poses.
// ============================================================================
__global__ __launch_bounds__(256) void posehead_kernel(
    const bf16* pf, const bf16* WP0, const float* pb0,
    const float* pb1, const float* pw1,
    float* o_plog, float* o_pprob, float* o_prmsd)
{
    extern __shared__ char smem_raw[];
    int w = threadIdx.x >> 5, lane = threadIdx.x & 31;
    bf16* act = (bf16*)smem_raw + (size_t)w * 16 * HID;
    int p0 = (blockIdx.x * 8 + w) * 16;

    v8f acc[16];
#pragma unroll
    for (int i = 0; i < 16; ++i) zero8(acc[i]);
    for (int k0 = 0; k0 < PF_LD; k0 += 32) {
        v16bf a = load_frag_A(pf + (size_t)p0 * PF_LD, PF_LD, k0, lane);
#pragma unroll
        for (int nt = 0; nt < 16; ++nt) {
            v16bf b = load_frag_B(WP0, PF_LD, nt * 16, k0, lane);
            acc[nt] = __builtin_amdgcn_wmma_f32_16x16x32_bf16(
                false, a, false, b, (short)0, acc[nt], false, false);
        }
    }
    epi16x256(acc, pb0, true, act, HID, lane);

    {
        int row = lane & 15, c = lane >> 4;
        float s = pb1[c];
        for (int k = 0; k < HID; ++k)
            s += (float)act[(size_t)row * HID + k] * pw1[(size_t)c * HID + k];
        int p = p0 + row;
        if (c == 0) { o_plog[p] = s; o_prmsd[p] = cexpm1(s); }
        else        { o_pprob[p] = s; }
    }
}

// ============================================================================
extern "C" void kernel_launch(void* const* d_in, const int* in_sizes, int n_in,
                              void* d_out, int out_size, void* d_ws, size_t ws_size,
                              hipStream_t stream)
{
    (void)in_sizes; (void)n_in; (void)out_size; (void)ws_size;

    const float* xs        = (const float*)d_in[0];
    const float* xn        = (const float*)d_in[1];
    const int*   pptr      = (const int*)d_in[2];
    const int*   frag_id   = (const int*)d_in[3];
    const float* stats     = (const float*)d_in[4];
    const float* ah0_b     = (const float*)d_in[5];
    const float* ah0_w     = (const float*)d_in[6];
    const float* ah1_b     = (const float*)d_in[7];
    const float* ah1_w     = (const float*)d_in[8];
    const float* attn_in_b = (const float*)d_in[9];
    const float* attn_in_w = (const float*)d_in[10];
    const float* attn_out_b= (const float*)d_in[11];
    const float* attn_out_w= (const float*)d_in[12];
    const float* fh0_b     = (const float*)d_in[13];
    const float* fh0_w     = (const float*)d_in[14];
    const float* fh1_b     = (const float*)d_in[15];
    const float* fh1_w     = (const float*)d_in[16];
    const float* poolq     = (const float*)d_in[17];
    const float* ph0_b     = (const float*)d_in[18];
    const float* ph0_w     = (const float*)d_in[19];
    const float* ph1_b     = (const float*)d_in[20];
    const float* ph1_w     = (const float*)d_in[21];
    const float* t0_b      = (const float*)d_in[22];
    const float* t0_w      = (const float*)d_in[23];
    const float* t1_b      = (const float*)d_in[24];
    const float* t1_w      = (const float*)d_in[25];
    const float* t2_b      = (const float*)d_in[26];
    const float* t2_w      = (const float*)d_in[27];

    char* ws = (char*)d_ws;
    bf16*  H2   = (bf16*)(ws + OFF_H2);
    float* FACC = (float*)(ws + OFF_FACC);
    float* FCNT = (float*)(ws + OFF_FCNT);
    bf16*  W0   = (bf16*)(ws + OFF_W0);
    bf16*  W1   = (bf16*)(ws + OFF_W1);
    bf16*  W2   = (bf16*)(ws + OFF_W2);
    bf16*  WA0  = (bf16*)(ws + OFF_WA0);
    bf16*  WF0  = (bf16*)(ws + OFF_WF0);
    bf16*  WK   = (bf16*)(ws + OFF_WK);
    bf16*  WV   = (bf16*)(ws + OFF_WV);
    bf16*  WP0  = (bf16*)(ws + OFF_WP0);
    bf16*  QM   = (bf16*)(ws + OFF_QM);
    bf16*  PF   = (bf16*)(ws + OFF_PF);

    float* out      = (float*)d_out;
    float* o_adlog  = out;
    float* o_aplddt = out + (size_t)N_ATOM;
    float* o_adisp  = out + (size_t)2 * N_ATOM;
    float* o_flog   = out + (size_t)3 * N_ATOM;
    float* o_fbad   = o_flog + NFRAG;
    float* o_frmsd  = o_fbad + NFRAG;
    float* o_plog   = o_frmsd + NFRAG;
    float* o_pprob  = o_plog + P_POSE;
    float* o_prmsd  = o_pprob + P_POSE;
    float* o_prfa   = o_prmsd + P_POSE;

    constexpr int TRUNK_SMEM = 128 * IN_DIM * 2;  // 176 KB (X tile / activations)
    constexpr int MLP_SMEM   = 8 * 16 * HID * 2;  // 64 KB
    constexpr int POSE_SMEM  = 3 * 128 * 256 * 2 + 16 * 128 * 4 + 16 * 128 * 2
                               + 4 * 256 * 4 + 128 * 4;  // 213504 B (< 320 KB WGP LDS)

    (void)hipFuncSetAttribute((const void*)trunk_kernel,
                              hipFuncAttributeMaxDynamicSharedMemorySize, TRUNK_SMEM);
    (void)hipFuncSetAttribute((const void*)frag_kernel,
                              hipFuncAttributeMaxDynamicSharedMemorySize, MLP_SMEM);
    (void)hipFuncSetAttribute((const void*)posehead_kernel,
                              hipFuncAttributeMaxDynamicSharedMemorySize, MLP_SMEM);
    (void)hipFuncSetAttribute((const void*)pose_kernel,
                              hipFuncAttributeMaxDynamicSharedMemorySize, POSE_SMEM);

    prep_kernel<<<1024, 256, 0, stream>>>(
        t0_w, t1_w, t2_w, ah0_w, fh0_w, attn_in_w, attn_in_b, poolq, ph0_w,
        W0, W1, W2, WA0, WF0, WK, WV, WP0, QM, FACC, FCNT);

    trunk_kernel<<<N_ATOM / 128, 256, TRUNK_SMEM, stream>>>(
        xs, xn, frag_id, t0_b, t1_b, t2_b, ah0_b, ah1_b, ah1_w,
        W0, W1, W2, WA0, H2, FACC, FCNT, o_adlog, o_aplddt, o_adisp);

    frag_kernel<<<NFRAG / 128, 256, MLP_SMEM, stream>>>(
        FACC, FCNT, WF0, fh0_b, fh1_b, fh1_w, o_flog, o_fbad, o_frmsd);

    pose_kernel<<<P_POSE, 256, POSE_SMEM, stream>>>(
        H2, o_adisp, pptr, stats, WK, WV, QM, attn_in_b, attn_out_w, attn_out_b,
        PF, o_prfa);

    posehead_kernel<<<P_POSE / 128, 256, MLP_SMEM, stream>>>(
        PF, WP0, ph0_b, ph1_b, ph1_w, o_plog, o_pprob, o_prmsd);
}
